// BasicBlock_26225070310076
// MI455X (gfx1250) — matrix operand order repaired and checked by reading, not compiled, not run on hardware
//
#include <hip/hip_runtime.h>
#include <stdint.h>

// ---------------------------------------------------------------------------
// CDNA5 (gfx1250) fused sparse-conv BasicBlock.
//  conv1 -> BN -> ReLU -> conv2 -> BN -> (+x) -> ReLU
// Conv = gather-GEMM-scatter using v_wmma_f32_16x16x32_bf16 (wave32).
//  - B (weight) tiles: Tensor Data Mover, double buffered, HW pad-swizzled.
//  - A (activation) tiles: bf16 rows gathered via GLOBAL_LOAD_ASYNC_TO_LDS,
//    mask applied by redirecting dead rows to a zero row (pure DMA staging).
// ---------------------------------------------------------------------------

typedef __attribute__((ext_vector_type(8)))  __bf16 v8bf;
typedef __attribute__((ext_vector_type(16))) __bf16 v16bf;
typedef __attribute__((ext_vector_type(8)))  float  v8f;
typedef __attribute__((ext_vector_type(4)))  unsigned int v4u;
typedef __attribute__((ext_vector_type(8)))  int v8i_;
typedef __attribute__((ext_vector_type(4)))  int v4i_;

// async_to_lds builtin operand types: int __vector(4) pointers in AS1 / AS3
typedef int v4i_gcc __attribute__((vector_size(16)));
typedef __attribute__((address_space(1))) v4i_gcc* gv4p;
typedef __attribute__((address_space(3))) v4i_gcc* lv4p;

#define C_DIM   128
#define KOFF    27
#define TILE_M  128
#define LDA     136                 // padded LDS row stride (ushorts): 272B -> lane-unique banks
#define A_BYTES (TILE_M * LDA * 2)  // 34816
#define B_BYTES (C_DIM * LDA * 2)   // 34816 per buffer, double buffered
#define RED_OFF (A_BYTES + 2 * B_BYTES)
#define SMEM_BYTES (A_BYTES + 2 * B_BYTES + 2 * C_DIM * 4)   // ~103 KB dynamic LDS (320KB/WGP)
#define EPSV 1e-5f

#if __has_builtin(__builtin_amdgcn_global_load_async_to_lds_b128)
#define HAVE_ASYNC_LDS 1
#else
#define HAVE_ASYNC_LDS 0
#pragma message("async_to_lds builtin unavailable: using synchronous A staging")
#endif

__device__ __forceinline__ uint16_t f2bf(float f) {
    uint32_t u = __float_as_uint(f);
    u += 0x7FFFu + ((u >> 16) & 1u);        // round-to-nearest-even
    return (uint16_t)(u >> 16);
}
__device__ __forceinline__ uint32_t pk2(float a, float b) {
    return (uint32_t)f2bf(a) | ((uint32_t)f2bf(b) << 16);
}

#if HAVE_ASYNC_LDS
__device__ __forceinline__ void async_cp16(const void* g, void* l) {
    __builtin_amdgcn_global_load_async_to_lds_b128((gv4p)g, (lv4p)l, 0, 0);
}
#endif

// --- TDM: async-load one 128x128 bf16 tile (row stride 128) from global into
//     LDS at lds_off, inserting 16B padding every 256B (-> 272B LDS row stride).
//     D# encoding per CDNA5 ISA (2D tensor).
__device__ __forceinline__ void tdm_load_w_tile(const uint16_t* gsrc, uint32_t lds_off) {
    uint64_t ga = (uint64_t)(uintptr_t)gsrc;
    v4u g0;
    g0[0] = 1u;                                                 // count=1 (valid), user mode
    g0[1] = lds_off;                                            // lds_addr (bytes)
    g0[2] = (uint32_t)ga;                                       // global_addr[31:0]
    g0[3] = ((uint32_t)(ga >> 32) & 0x01FFFFFFu) | (2u << 30);  // global_addr[56:32] | type=2
    v8i_ g1;
    g1[0] = (1 << 16)        // data_size = 2 bytes
          | (1 << 20)        // pad_enable
          | (5 << 22)        // pad_interval code 5 -> every 256 bytes
          | (3 << 25);       // pad_amount  code 3 -> 4 DWORDs = 16 bytes
    g1[1] = (C_DIM & 0xFFFF) << 16;   // tensor_dim0[15:0] @ bit48
    g1[2] = (C_DIM << 16);            // tensor_dim0 hi=0 | tensor_dim1[15:0] @ bit80
    g1[3] = (C_DIM << 16);            // tensor_dim1 hi=0 | tile_dim0 @ bit112
    g1[4] = C_DIM;                    // tile_dim1 | tile_dim2=0
    g1[5] = C_DIM;                    // tensor_dim0_stride[31:0]
    g1[6] = 0;
    g1[7] = 0;
    v4i_ z4 = {0, 0, 0, 0};           // groups 2/3 unused (2D tensor)
    v8i_ z8 = {0, 0, 0, 0, 0, 0, 0, 0};
    __builtin_amdgcn_tensor_load_to_lds(g0, g1, z4, z4, z8, 0);
}

// --- W -> bf16, transposed to [k][n_out][c_in] so the TDM B tile is contiguous ---
__global__ void wt_prep(const float* __restrict__ W1, const float* __restrict__ W2,
                        uint16_t* __restrict__ Wt1, uint16_t* __restrict__ Wt2) {
    int i = blockIdx.x * blockDim.x + threadIdx.x;
    const int per = KOFF * C_DIM * C_DIM;
    if (i >= 2 * per) return;
    const float* W  = (i < per) ? W1 : W2;
    uint16_t*    Wt = (i < per) ? Wt1 : Wt2;
    int j = (i < per) ? i : i - per;
    int k = j / (C_DIM * C_DIM);
    int r = j - k * C_DIM * C_DIM;
    int n = r / C_DIM;              // out channel (row of Wt)
    int c = r - n * C_DIM;          // in channel  (col of Wt)
    Wt[j] = f2bf(W[k * C_DIM * C_DIM + c * C_DIM + n]);
}

// --- f32 -> packed bf16 (activations), grid over total/4 ---
__global__ void to_bf16(const float* __restrict__ S, uint16_t* __restrict__ D, int total4) {
    int i = blockIdx.x * blockDim.x + threadIdx.x;
    if (i >= total4) return;
    float4 v = ((const float4*)S)[i];
    ((uint2*)D)[i] = make_uint2(pk2(v.x, v.y), pk2(v.z, v.w));
}

__global__ void zfill(uint16_t* __restrict__ p, int n) {
    int i = blockIdx.x * blockDim.x + threadIdx.x;
    if (i < n) p[i] = 0;
}

// --- gather-GEMM sparse conv: H[n,:] = sum_k mask[k,n] * Xbf[kmap[k,n],:] @ W[k]
//     plus per-workgroup column partial sums of H and H^2 (for BN).
__global__ __launch_bounds__(256)
void sconv_wmma(const uint16_t* __restrict__ Xbf, const uint16_t* __restrict__ zrow,
                const uint16_t* __restrict__ Wt,
                const int* __restrict__ kmap, const int* __restrict__ mask,
                float* __restrict__ H, float* __restrict__ psum,
                float* __restrict__ psq, int Ntot) {
    extern __shared__ char smem[];
    uint16_t* As   = (uint16_t*)smem;
    uint16_t* Bbuf[2];
    Bbuf[0] = (uint16_t*)(smem + A_BYTES);
    Bbuf[1] = (uint16_t*)(smem + A_BYTES + B_BYTES);
    float* red  = (float*)(smem + RED_OFF);
    float* red2 = red + C_DIM;

    const int t  = threadIdx.x;
    const int n0 = blockIdx.x * TILE_M;

    if (t < C_DIM) { red[t] = 0.f; red2[t] = 0.f; }

    const int lane = t & 31;
    const int wave = t >> 5;                 // 8 waves (wave-uniform)
    const int wm = wave >> 1, wn = wave & 1; // 4x2 wave grid
    const int rowBase = wm * 32, colBase = wn * 64;
    const int half = lane >> 4, l16 = lane & 15;

    v8f acc[8];
    const v8f vzero = {0.f,0.f,0.f,0.f,0.f,0.f,0.f,0.f};
#pragma unroll
    for (int i = 0; i < 8; i++) acc[i] = vzero;

    // Prime the TDM pipe: B tile for k=0 into buffer 0 (wave-uniform branch;
    // TDM ignores EXEC, so only wave 0 may reach the instruction).
    if (wave == 0) {
        tdm_load_w_tile(Wt, (uint32_t)A_BYTES);
    }

    for (int k = 0; k < KOFF; k++) {
        __syncthreads();   // everyone done computing on LDS from iteration k-1

        // ---- stage A: masked bf16 row gather, async global -> LDS ----
#if HAVE_ASYNC_LDS
        {
            // per wave: 16 rows; per issue: 2 rows x 16 granules of 16B
            const int arow = wave * 16 + (lane >> 4);
            const int aoff = (lane & 15) * 16;   // byte offset in 256B row
#pragma unroll
            for (int rr = 0; rr < 8; rr++) {
                const int r = arow + rr * 2;
                const int n = n0 + r;
                const uint16_t* src = zrow;
                if (n < Ntot) {
                    const int idxv = kmap[k * Ntot + n];
                    const int m    = mask[k * Ntot + n];
                    if (m) src = Xbf + (size_t)idxv * C_DIM;
                }
                async_cp16((const char*)src + aoff,
                           (char*)As + r * (LDA * 2) + aoff);
            }
            asm volatile("s_wait_asynccnt 0" ::: "memory");
        }
#else
        {
            const int srow = t >> 1;   // 2 threads per row
            const int sseg = t & 1;
            const int n = n0 + srow;
            const uint16_t* src = zrow;
            if (n < Ntot) {
                const int idxv = kmap[k * Ntot + n];
                const int m    = mask[k * Ntot + n];
                if (m) src = Xbf + (size_t)idxv * C_DIM;
            }
            const uint4* s4 = (const uint4*)(src + sseg * 64);
            uint4* d4 = (uint4*)(As + srow * LDA + sseg * 64);
#pragma unroll
            for (int j = 0; j < 8; j++) d4[j] = s4[j];
        }
#endif
        if (k + 1 < KOFF && t < TILE_M && n0 + t < Ntot) {
            __builtin_prefetch(&kmap[(k + 1) * Ntot + n0 + t], 0, 1);
            __builtin_prefetch(&mask[(k + 1) * Ntot + n0 + t], 0, 1);
        }

        // ---- TDM pipeline: kick k+1 into the other buffer, then confirm k ----
        if (wave == 0) {
            if (k + 1 < KOFF) {
                tdm_load_w_tile(Wt + (size_t)(k + 1) * C_DIM * C_DIM,
                                (uint32_t)(A_BYTES + ((k + 1) & 1) * B_BYTES));
                // TDM ops retire in order: <=1 outstanding proves tile k landed.
                __builtin_amdgcn_s_wait_tensorcnt(1);
            } else {
                __builtin_amdgcn_s_wait_tensorcnt(0);
            }
        }
        __syncthreads();   // A staged + B tile k visible to all waves

        const uint16_t* Bs = Bbuf[k & 1];

        // ---- WMMA: 4 K-chunks of 32, 2x4 output tiles per wave ----
#pragma unroll
        for (int kc = 0; kc < 4; kc++) {
            const int ab = kc * 32;
            v16bf a[2], b[4];
#pragma unroll
            for (int ti = 0; ti < 2; ti++) {
                // A 16x32 bf16: lanes 0-15 hold K {0..7,16..23}, lanes 16-31 {8..15,24..31}
                const uint16_t* p = As + (rowBase + ti * 16 + l16) * LDA + ab + half * 8;
                v8bf lo = *(const v8bf*)p;
                v8bf hi = *(const v8bf*)(p + 16);
                a[ti] = __builtin_shufflevector(lo, hi, 0,1,2,3,4,5,6,7,8,9,10,11,12,13,14,15);
            }
#pragma unroll
            for (int tj = 0; tj < 4; tj++) {
                // B 32x16 bf16: lane = N column, lanes 0-15 K {0..7,16..23}, lanes 16-31 {8..15,24..31}
                const uint16_t* p = Bs + (colBase + tj * 16 + l16) * LDA + ab + half * 8;
                v8bf lo = *(const v8bf*)p;
                v8bf hi = *(const v8bf*)(p + 16);
                b[tj] = __builtin_shufflevector(lo, hi, 0,1,2,3,4,5,6,7,8,9,10,11,12,13,14,15);
            }
#pragma unroll
            for (int ti = 0; ti < 2; ti++)
#pragma unroll
                for (int tj = 0; tj < 4; tj++)
                    acc[ti * 4 + tj] = __builtin_amdgcn_wmma_f32_16x16x32_bf16(
                        false, a[ti], false, b[tj], (short)0, acc[ti * 4 + tj], false, false);
        }
    }

    // ---- store H tile + BN partial sums (invalid rows are all-zero -> no-op) ----
#pragma unroll
    for (int ti = 0; ti < 2; ti++) {
#pragma unroll
        for (int tj = 0; tj < 4; tj++) {
            const int gc = colBase + tj * 16 + l16;
            float s = 0.f, s2 = 0.f;
#pragma unroll
            for (int v = 0; v < 8; v++) {
                // C/D layout: VGPR v -> M = v + 8*half, lane%16 -> N
                const int gr = n0 + rowBase + ti * 16 + v + 8 * half;
                const float val = acc[ti * 4 + tj][v];
                s += val; s2 += val * val;
                if (gr < Ntot) H[(size_t)gr * C_DIM + gc] = val;
            }
            atomicAdd(&red[gc], s);     // ds_add_f32
            atomicAdd(&red2[gc], s2);
        }
    }
    __syncthreads();
    if (t < C_DIM) {
        psum[(size_t)blockIdx.x * C_DIM + t] = red[t];
        psq [(size_t)blockIdx.x * C_DIM + t] = red2[t];
    }
}

// --- deterministic BN stat reduction: scale/shift per channel ---
__global__ void bn_reduce(const float* __restrict__ psum, const float* __restrict__ psq,
                          const float* __restrict__ gamma, const float* __restrict__ beta,
                          float* __restrict__ scsh, int nWG, int Ntot) {
    int c = threadIdx.x;
    float m = 0.f, q = 0.f;
    for (int w = 0; w < nWG; w++) {
        m += psum[(size_t)w * C_DIM + c];
        q += psq [(size_t)w * C_DIM + c];
    }
    const float inv_n = 1.f / (float)Ntot;
    const float mean = m * inv_n;
    const float var  = q * inv_n - mean * mean;
    const float s = gamma[c] * rsqrtf(var + EPSV);
    scsh[c] = s;
    scsh[C_DIM + c] = beta[c] - mean * s;
}

// --- y(bf16) = relu(bn(h)) : feeds conv2's async gather directly ---
__global__ void bn_relu_bf16(const float* __restrict__ H, const float* __restrict__ scsh,
                             uint16_t* __restrict__ Y, int total4) {
    int i = blockIdx.x * blockDim.x + threadIdx.x;
    if (i >= total4) return;
    const int c = (i * 4) & (C_DIM - 1);
    float4 h = ((const float4*)H)[i];
    float4 o;
    o.x = fmaxf(h.x * scsh[c+0] + scsh[C_DIM+c+0], 0.f);
    o.y = fmaxf(h.y * scsh[c+1] + scsh[C_DIM+c+1], 0.f);
    o.z = fmaxf(h.z * scsh[c+2] + scsh[C_DIM+c+2], 0.f);
    o.w = fmaxf(h.w * scsh[c+3] + scsh[C_DIM+c+3], 0.f);
    ((uint2*)Y)[i] = make_uint2(pk2(o.x, o.y), pk2(o.z, o.w));
}

// --- out = relu(bn(h) + x), in place on H ---
__global__ void bn_add_relu(float* __restrict__ H, const float* __restrict__ X,
                            const float* __restrict__ scsh, int total4) {
    int i = blockIdx.x * blockDim.x + threadIdx.x;
    if (i >= total4) return;
    const int c = (i * 4) & (C_DIM - 1);
    float4 h = ((float4*)H)[i];
    float4 x = ((const float4*)X)[i];
    float4 o;
    o.x = fmaxf(h.x * scsh[c+0] + scsh[C_DIM+c+0] + x.x, 0.f);
    o.y = fmaxf(h.y * scsh[c+1] + scsh[C_DIM+c+1] + x.y, 0.f);
    o.z = fmaxf(h.z * scsh[c+2] + scsh[C_DIM+c+2] + x.z, 0.f);
    o.w = fmaxf(h.w * scsh[c+3] + scsh[C_DIM+c+3] + x.w, 0.f);
    ((float4*)H)[i] = o;
}

extern "C" void kernel_launch(void* const* d_in, const int* in_sizes, int n_in,
                              void* d_out, int out_size, void* d_ws, size_t ws_size,
                              hipStream_t stream) {
    (void)n_in; (void)out_size; (void)ws_size;
    const float* x     = (const float*)d_in[0];
    const float* W1    = (const float*)d_in[1];
    const float* g1    = (const float*)d_in[2];
    const float* b1    = (const float*)d_in[3];
    const float* W2    = (const float*)d_in[4];
    const float* g2    = (const float*)d_in[5];
    const float* b2    = (const float*)d_in[6];
    const int*   kmap1 = (const int*)d_in[7];
    const int*   kmap2 = (const int*)d_in[8];
    const int*   mask1 = (const int*)d_in[9];
    const int*   mask2 = (const int*)d_in[10];
    float* out = (float*)d_out;

    const int Ntot = in_sizes[0] / C_DIM;
    const int nWG  = (Ntot + TILE_M - 1) / TILE_M;

    // workspace carve-up (256B aligned)
    char* ws = (char*)d_ws; size_t off = 0;
    auto alloc = [&](size_t bytes) -> char* {
        char* p = ws + off; off += (bytes + 255) & ~(size_t)255; return p;
    };
    uint16_t* Wt1  = (uint16_t*)alloc((size_t)KOFF * C_DIM * C_DIM * 2);
    uint16_t* Wt2  = (uint16_t*)alloc((size_t)KOFF * C_DIM * C_DIM * 2);
    uint16_t* xbf  = (uint16_t*)alloc((size_t)Ntot * C_DIM * 2);
    uint16_t* y1bf = (uint16_t*)alloc((size_t)Ntot * C_DIM * 2);
    uint16_t* zrow = (uint16_t*)alloc(C_DIM * 2);
    float* psum    = (float*)alloc((size_t)nWG * C_DIM * 4);
    float* psq     = (float*)alloc((size_t)nWG * C_DIM * 4);
    float* scsh1   = (float*)alloc(2 * C_DIM * 4);
    float* scsh2   = (float*)alloc(2 * C_DIM * 4);

    (void)hipFuncSetAttribute((const void*)sconv_wmma,
                              hipFuncAttributeMaxDynamicSharedMemorySize, SMEM_BYTES);

    const int total4 = Ntot * C_DIM / 4;
    const int prepTot = 2 * KOFF * C_DIM * C_DIM;
    wt_prep<<<(prepTot + 255) / 256, 256, 0, stream>>>(W1, W2, Wt1, Wt2);
    to_bf16<<<(total4 + 255) / 256, 256, 0, stream>>>(x, xbf, total4);
    zfill<<<1, C_DIM, 0, stream>>>(zrow, C_DIM);

    // conv1 -> d_out, then y1(bf16) = relu(bn1(d_out))
    sconv_wmma<<<nWG, 256, SMEM_BYTES, stream>>>(xbf, zrow, Wt1, kmap1, mask1, out, psum, psq, Ntot);
    bn_reduce<<<1, C_DIM, 0, stream>>>(psum, psq, g1, b1, scsh1, nWG, Ntot);
    bn_relu_bf16<<<(total4 + 255) / 256, 256, 0, stream>>>(out, scsh1, y1bf, total4);

    // conv2 -> d_out, then out = relu(bn2(d_out) + x) in place
    sconv_wmma<<<nWG, 256, SMEM_BYTES, stream>>>(y1bf, zrow, Wt2, kmap2, mask2, out, psum, psq, Ntot);
    bn_reduce<<<1, C_DIM, 0, stream>>>(psum, psq, g2, b2, scsh2, nWG, Ntot);
    bn_add_relu<<<(total4 + 255) / 256, 256, 0, stream>>>(out, x, scsh2, total4);
}
